// Seq2Seq_4234837753867
// MI455X (gfx1250) — matrix operand order repaired and checked by reading, not compiled, not run on hardware
//
#include <hip/hip_runtime.h>

#define HID   512
#define BATCH 32
#define S_ENC 128
#define T_DEC 64
#define VOCAB 32000
#define NSTEP (S_ENC + T_DEC - 1)   // 191
#define NBLK  32                    // blocks in persistent scan

// LDS B-panel: 64 rows, stride padded to 520 ushorts (1040B = 260 dwords,
// 260 mod 64 = 4 -> lanes 0..15 (stride 4 banks, 4-dword loads) hit all 64
// banks with no conflict; 1040 is 16B-aligned for ds_load_b128.
#define BPAN_STRIDE 520

typedef __attribute__((ext_vector_type(16))) __bf16 v16bf;
typedef __attribute__((ext_vector_type(8)))  __bf16 v8bf;
typedef __attribute__((ext_vector_type(8)))  float  v8f;

__device__ __forceinline__ unsigned short f2bf(float f) {
    union { float f; unsigned u; } v; v.f = f;
    unsigned r = v.u + 0x7FFFu + ((v.u >> 16) & 1u);   // round-to-nearest-even
    return (unsigned short)(r >> 16);
}

__device__ __forceinline__ float sigmf(float x) { return 1.0f / (1.0f + __expf(-x)); }
__device__ __forceinline__ float tanhfast(float x) {
    float e2 = __expf(2.0f * x);
    return (e2 - 1.0f) / (e2 + 1.0f);
}

// A tile 16x32 bf16 from row-major [*, ld]: lane = M row; lanes 0-15 hold
// K {k0..k0+7, k0+16..k0+23}; lanes 16-31 hold K {k0+8.., k0+24..}.
__device__ __forceinline__ v16bf load_tileA(const unsigned short* __restrict__ M,
                                            int ld, int row0, int k0, int lane) {
    int r = row0 + (lane & 15);
    int kadd = (lane < 16) ? 0 : 8;
    const unsigned short* p = M + r * ld + k0 + kadd;
    v8bf lo = *(const v8bf*)p;
    v8bf hi = *(const v8bf*)(p + 16);
    v16bf a;
#pragma unroll
    for (int i = 0; i < 8; ++i) { a[i] = lo[i]; a[i + 8] = hi[i]; }
    return a;
}

// B tile from the padded LDS panel (two ds_load_b128 / lane): lane = N row of
// the panel, lanes 0-15 hold K {k0..k0+7, k0+16..k0+23}; lanes 16-31 hold K {k0+8.., k0+24..}.
__device__ __forceinline__ v16bf load_tileB_lds(const unsigned short* sB,
                                                int nloc0, int k0, int lane) {
    int n = nloc0 + (lane & 15);
    int kk = k0 + ((lane < 16) ? 0 : 8);
    const unsigned short* p = sB + n * BPAN_STRIDE + kk;
    v8bf lo = *(const v8bf*)p;
    v8bf hi = *(const v8bf*)(p + 16);
    v16bf b;
#pragma unroll
    for (int i = 0; i < 8; ++i) { b[i] = lo[i]; b[i + 8] = hi[i]; }
    return b;
}

__device__ __forceinline__ v8f wmma_bf16(v16bf a, v16bf b, v8f c) {
    return __builtin_amdgcn_wmma_f32_16x16x32_bf16(false, a, false, b,
                                                   (short)0, c, false, false);
}

// Async-stage a 64-row x 512-K bf16 panel (contiguous global rows n0..n0+63)
// into padded LDS. 4096 16B chunks, 128 threads -> 32 async b128 issues each.
__device__ __forceinline__ void stage_rows64(unsigned short* sB,
                                             const unsigned short* W,
                                             int n0, int tid) {
    unsigned sbase = (unsigned)(size_t)sB;
    unsigned long long wb = (unsigned long long)(size_t)W;
#pragma unroll 4
    for (int i = 0; i < 32; ++i) {
        int q = i * 128 + tid;
        int r = q >> 6;
        int cch = q & 63;
        unsigned laddr = sbase + (unsigned)(r * (BPAN_STRIDE * 2) + cch * 16);
        unsigned goff  = (unsigned)((n0 + r) * (HID * 2) + cch * 16);
        asm volatile("global_load_async_to_lds_b128 %0, %1, %2"
                     :: "v"(laddr), "v"(goff), "s"(wb) : "memory");
    }
}

// Same, but local row lr = g*16+r maps to global gate row g*512 + n0 + r.
__device__ __forceinline__ void stage_gate_panel(unsigned short* sB,
                                                 const unsigned short* W,
                                                 int n0, int tid) {
    unsigned sbase = (unsigned)(size_t)sB;
    unsigned long long wb = (unsigned long long)(size_t)W;
#pragma unroll 4
    for (int i = 0; i < 32; ++i) {
        int q = i * 128 + tid;
        int lr = q >> 6;                 // 0..63 local row
        int cch = q & 63;
        int g = lr >> 4, r = lr & 15;
        unsigned laddr = sbase + (unsigned)(lr * (BPAN_STRIDE * 2) + cch * 16);
        unsigned goff  = (unsigned)((g * HID + n0 + r) * (HID * 2) + cch * 16);
        asm volatile("global_load_async_to_lds_b128 %0, %1, %2"
                     :: "v"(laddr), "v"(goff), "s"(wb) : "memory");
    }
}

// ---------------- utility kernels ----------------

__global__ void k_cvt_bf16(const float* __restrict__ src, unsigned short* __restrict__ dst, int n) {
    int i = blockIdx.x * blockDim.x + threadIdx.x;
    if (i < n) dst[i] = f2bf(src[i]);
}

__global__ void k_bias_sum(const float* __restrict__ a, const float* __restrict__ b,
                           float* __restrict__ o, int n) {
    int i = blockIdx.x * blockDim.x + threadIdx.x;
    if (i < n) o[i] = a[i] + b[i];
}

__global__ void k_zero_u16(unsigned short* __restrict__ p, int n) {
    int i = blockIdx.x * blockDim.x + threadIdx.x;
    if (i < n) p[i] = 0;
}
__global__ void k_zero_u32(unsigned* __restrict__ p, int n) {
    int i = blockIdx.x * blockDim.x + threadIdx.x;
    if (i < n) p[i] = 0u;
}

// Gather embeddings time-major: xout[(s*32+b)*512+k] = bf16(emb[tok[b*S+s]*512+k]).
// If aproj != null, also write s==0 rows into A_proj rows (b*64 + 0).
__global__ void k_gather(const int* __restrict__ tok, const float* __restrict__ emb,
                         unsigned short* __restrict__ xout, unsigned short* __restrict__ aproj,
                         int S, int n) {
    int idx = blockIdx.x * blockDim.x + threadIdx.x;
    if (idx >= n) return;
    int k   = idx & 511;
    int row = idx >> 9;
    int s   = row >> 5;
    int b   = row & 31;
    int t   = tok[b * S + s];
    unsigned short v = f2bf(emb[t * HID + k]);
    xout[idx] = v;
    if (aproj != nullptr && s == 0) aproj[(b * T_DEC) * HID + k] = v;
}

// ---------------- persistent LSTM scan (encoder + decoder) ----------------
// 32 blocks x 128 threads. Block bid owns hidden slice j in [bid*16, bid*16+16).
// LDS holds this slice's 64 gate-rows of Wih and Whh for the current phase
// (async-staged once per phase). Waves: wid&1 = M tile, wid>>1 = K half.
// Cell state c lives in VGPRs of the khalf==0 waves for all 191 steps.
// Steps separated by a device-wide atomic barrier (monotonic target).
__global__ __launch_bounds__(128)
void k_scan(const unsigned short* __restrict__ x_enc,   // [128,32,512] bf16
            const unsigned short* __restrict__ x_dec,   // [64,32,512] bf16
            const unsigned short* __restrict__ encWih,
            const unsigned short* __restrict__ encWhh,
            const unsigned short* __restrict__ decWih,
            const unsigned short* __restrict__ decWhh,
            const float* __restrict__ enc_bias,         // [2048]
            const float* __restrict__ dec_bias,         // [2048]
            unsigned short* __restrict__ hb0,           // [32,512] bf16 ping
            unsigned short* __restrict__ hb1,           // [32,512] bf16 pong
            unsigned short* __restrict__ aproj,         // [2048,512] bf16
            unsigned* __restrict__ ctr) {               // global barrier counter
    __shared__ unsigned short sWih[64 * BPAN_STRIDE];   // 66.6 KB
    __shared__ unsigned short sWhh[64 * BPAN_STRIDE];   // 66.6 KB
    __shared__ float sred[2][4][256];                   // 8 KB K-partials

    const int tid   = threadIdx.x;
    const int lane  = tid & 31;
    const int wid   = tid >> 5;
    const int mt    = wid & 1;
    const int khalf = wid >> 1;
    const int m0    = mt * 16;
    const int n0    = (int)blockIdx.x * 16;
    const int kbase = khalf * 256;

    unsigned short* hbuf[2] = { hb0, hb1 };
    const float* bias = enc_bias;

    v8f creg;                                   // cell state (khalf==0 waves)
#pragma unroll
    for (int v = 0; v < 8; ++v) creg[v] = 0.0f;

    int pp = 0;
    for (int step = 0; step < NSTEP; ++step) {
        const unsigned short* xin;
        int t;
        if (step < S_ENC) { xin = x_enc + (size_t)step * BATCH * HID; t = 0; }
        else              { t = step - (S_ENC - 1); xin = x_dec + (size_t)t * BATCH * HID; }

        if (step == 0 || step == S_ENC) {       // stage phase weights into LDS
            const unsigned short* Wih = (step == 0) ? encWih : decWih;
            const unsigned short* Whh = (step == 0) ? encWhh : decWhh;
            bias = (step == 0) ? enc_bias : dec_bias;
            stage_gate_panel(sWih, Wih, n0, tid);
            stage_gate_panel(sWhh, Whh, n0, tid);
            asm volatile("s_wait_asynccnt 0x0" ::: "memory");
            __syncthreads();
        }

        v8f acc[4];
#pragma unroll
        for (int g = 0; g < 4; ++g) {
            float bv = (khalf == 0) ? bias[g * HID + n0 + (lane & 15)] : 0.0f;
#pragma unroll
            for (int v = 0; v < 8; ++v) acc[g][v] = bv;
        }

        const unsigned short* hin = hbuf[pp];
        // software pipeline: A tiles (h and x) fetched one k-iter ahead
        v16bf ah = load_tileA(hin, HID, m0, kbase, lane);
        v16bf ax = load_tileA(xin, HID, m0, kbase, lane);
#pragma unroll
        for (int ki = 0; ki < 8; ++ki) {
            int k0 = kbase + ki * 32;
            v16bf ah_n = ah, ax_n = ax;
            if (ki < 7) {
                ah_n = load_tileA(hin, HID, m0, k0 + 32, lane);
                ax_n = load_tileA(xin, HID, m0, k0 + 32, lane);
            }
#pragma unroll
            for (int g = 0; g < 4; ++g) {
                v16bf bh = load_tileB_lds(sWhh, g * 16, k0, lane);
                acc[g] = wmma_bf16(ah, bh, acc[g]);
                v16bf bx = load_tileB_lds(sWih, g * 16, k0, lane);
                acc[g] = wmma_bf16(ax, bx, acc[g]);
            }
            ah = ah_n; ax = ax_n;
        }

        // combine K halves via LDS
        if (khalf == 1) {
#pragma unroll
            for (int g = 0; g < 4; ++g)
#pragma unroll
                for (int v = 0; v < 8; ++v)
                    sred[mt][g][lane * 8 + v] = acc[g][v];
        }
        __syncthreads();
        if (khalf == 0) {
            const int j     = n0 + (lane & 15);
            const int mbase = m0 + ((lane < 16) ? 0 : 8);
            unsigned short* hout = hbuf[pp ^ 1];
#pragma unroll
            for (int v = 0; v < 8; ++v) {
                float gi = acc[0][v] + sred[mt][0][lane * 8 + v];
                float gf = acc[1][v] + sred[mt][1][lane * 8 + v];
                float gg = acc[2][v] + sred[mt][2][lane * 8 + v];
                float go = acc[3][v] + sred[mt][3][lane * 8 + v];
                float iv = sigmf(gi);
                float fv = sigmf(gf);
                float gv = tanhfast(gg);
                float ov = sigmf(go);
                float cn = fv * creg[v] + iv * gv;
                creg[v] = cn;
                float hv = ov * tanhfast(cn);
                unsigned short hbv = f2bf(hv);
                int b = mbase + v;
                hout[b * HID + j] = hbv;
                if (step >= S_ENC) aproj[(b * T_DEC + t) * HID + j] = hbv;
            }
        }

        // device-wide barrier: release h stores, arrive, spin, acquire
        __threadfence();
        __syncthreads();
        if (tid == 0) {
            atomicAdd(ctr, 1u);
            unsigned target = (unsigned)NBLK * (unsigned)(step + 1);
            while (__hip_atomic_load(ctr, __ATOMIC_RELAXED, __HIP_MEMORY_SCOPE_AGENT) < target) {
                __builtin_amdgcn_s_sleep(2);
            }
        }
        __syncthreads();
        __threadfence();
        pp ^= 1;
    }
}

// ---------------- projection: out = relu(A @ Wout^T + bout), f32 logits ----------------
// Block = 4 waves sharing one 64-column B panel (async-staged to LDS, all K),
// each wave a different 16-row M tile. Grid ordered M-fastest so the 32 blocks
// of one N-panel run adjacently (keeps the Wout slice hot in L2).
__global__ __launch_bounds__(128)
void k_proj(const unsigned short* __restrict__ A,   // [2048,512] bf16
            const unsigned short* __restrict__ W,   // [32000,512] bf16
            const float* __restrict__ bout,         // [32000]
            float* __restrict__ out) {              // [2048,32000]
    __shared__ unsigned short sB[64 * BPAN_STRIDE];  // ~66.6 KB
    const int tid  = threadIdx.x;
    const int lane = tid & 31;
    const int wid  = tid >> 5;
    const int n0   = (blockIdx.x >> 5) * 64;                  // 0..31936
    const int m0   = (((blockIdx.x & 31) << 2) | wid) * 16;   // 0..2032

    stage_rows64(sB, W, n0, tid);
    asm volatile("s_wait_asynccnt 0x0" ::: "memory");
    __syncthreads();

    v8f acc[4];
#pragma unroll
    for (int g = 0; g < 4; ++g)
#pragma unroll
        for (int v = 0; v < 8; ++v) acc[g][v] = 0.0f;

    for (int k0 = 0; k0 < HID; k0 += 32) {
        v16bf a = load_tileA(A, HID, m0, k0, lane);
        v16bf b0 = load_tileB_lds(sB, 0,  k0, lane);
        v16bf b1 = load_tileB_lds(sB, 16, k0, lane);
        v16bf b2 = load_tileB_lds(sB, 32, k0, lane);
        v16bf b3 = load_tileB_lds(sB, 48, k0, lane);
        acc[0] = wmma_bf16(a, b0, acc[0]);
        acc[1] = wmma_bf16(a, b1, acc[1]);
        acc[2] = wmma_bf16(a, b2, acc[2]);
        acc[3] = wmma_bf16(a, b3, acc[3]);
    }

    const int mbase = m0 + ((lane < 16) ? 0 : 8);
#pragma unroll
    for (int g = 0; g < 4; ++g) {
        int   n  = n0 + g * 16 + (lane & 15);
        float bo = bout[n];
#pragma unroll
        for (int v = 0; v < 8; ++v) {
            float val = fmaxf(acc[g][v] + bo, 0.0f);
            out[(size_t)(mbase + v) * VOCAB + n] = val;
        }
    }
}

// ---------------- logsumexp over vocab per row ----------------
__global__ __launch_bounds__(256)
void k_rowlse(const float* __restrict__ x, float* __restrict__ lse) {
    __shared__ float red[256];
    const int tid = threadIdx.x;
    const float* p = x + (size_t)blockIdx.x * VOCAB;
    float m = -1e30f;
    for (int i = tid; i < VOCAB; i += 256) m = fmaxf(m, p[i]);
    red[tid] = m; __syncthreads();
    for (int s = 128; s > 0; s >>= 1) {
        if (tid < s) red[tid] = fmaxf(red[tid], red[tid + s]);
        __syncthreads();
    }
    m = red[0]; __syncthreads();
    float sum = 0.0f;
    for (int i = tid; i < VOCAB; i += 256) sum += __expf(p[i] - m);
    red[tid] = sum; __syncthreads();
    for (int s = 128; s > 0; s >>= 1) {
        if (tid < s) red[tid] += red[tid + s];
        __syncthreads();
    }
    if (tid == 0) lse[blockIdx.x] = m + __logf(red[0]);
}

// ---------------- final: x - rowLSE - colLSE (batch-axis logsumexp) ----------------
__global__ __launch_bounds__(256)
void k_final(float* __restrict__ x, const float* __restrict__ rowlse) {
    int idx = blockIdx.x * blockDim.x + threadIdx.x;
    if (idx >= T_DEC * VOCAB) return;
    int t = idx / VOCAB;
    int v = idx - t * VOCAB;
    float r[BATCH];
    float m = -1e30f;
#pragma unroll
    for (int b = 0; b < BATCH; ++b) {
        float val = x[(size_t)(b * T_DEC + t) * VOCAB + v] - rowlse[b * T_DEC + t];
        r[b] = val;
        m = fmaxf(m, val);
    }
    float s = 0.0f;
#pragma unroll
    for (int b = 0; b < BATCH; ++b) s += __expf(r[b] - m);
    float l = m + __logf(s);
#pragma unroll
    for (int b = 0; b < BATCH; ++b)
        x[(size_t)(b * T_DEC + t) * VOCAB + v] = r[b] - l;
}

// ---------------- host orchestration ----------------
extern "C" void kernel_launch(void* const* d_in, const int* in_sizes, int n_in,
                              void* d_out, int out_size, void* d_ws, size_t ws_size,
                              hipStream_t stream) {
    (void)in_sizes; (void)n_in; (void)out_size; (void)ws_size;
    const int*   inp     = (const int*)d_in[0];    // [32,128]
    const int*   tar     = (const int*)d_in[1];    // [32,64]
    const float* enc_emb = (const float*)d_in[2];
    const float* dec_emb = (const float*)d_in[3];
    const float* enc_Wih = (const float*)d_in[4];
    const float* enc_Whh = (const float*)d_in[5];
    const float* enc_bih = (const float*)d_in[6];
    const float* enc_bhh = (const float*)d_in[7];
    const float* dec_Wih = (const float*)d_in[8];
    const float* dec_Whh = (const float*)d_in[9];
    const float* dec_bih = (const float*)d_in[10];
    const float* dec_bhh = (const float*)d_in[11];
    const float* Wout    = (const float*)d_in[12];
    const float* bout    = (const float*)d_in[13];
    float* out = (float*)d_out;

    // workspace carve-up (256B aligned)
    unsigned char* wsp = (unsigned char*)d_ws;
    auto alloc = [&](size_t bytes) -> void* {
        void* p = wsp;
        wsp += (bytes + 255) & ~(size_t)255;
        return p;
    };
    const int WG = 4 * HID * HID;            // 2048*512
    unsigned short* encWih_b = (unsigned short*)alloc((size_t)WG * 2);
    unsigned short* encWhh_b = (unsigned short*)alloc((size_t)WG * 2);
    unsigned short* decWih_b = (unsigned short*)alloc((size_t)WG * 2);
    unsigned short* decWhh_b = (unsigned short*)alloc((size_t)WG * 2);
    unsigned short* Wout_b   = (unsigned short*)alloc((size_t)VOCAB * HID * 2);
    float*          enc_bias = (float*)alloc(4 * HID * 4);
    float*          dec_bias = (float*)alloc(4 * HID * 4);
    unsigned short* x_enc    = (unsigned short*)alloc((size_t)S_ENC * BATCH * HID * 2);
    unsigned short* x_dec    = (unsigned short*)alloc((size_t)T_DEC * BATCH * HID * 2);
    unsigned short* aproj    = (unsigned short*)alloc((size_t)BATCH * T_DEC * HID * 2);
    unsigned short* hb0      = (unsigned short*)alloc((size_t)BATCH * HID * 2);
    unsigned short* hb1      = (unsigned short*)alloc((size_t)BATCH * HID * 2);
    float*          rowlse   = (float*)alloc((size_t)BATCH * T_DEC * 4);
    unsigned*       ctr      = (unsigned*)alloc(256);

    // 1) weight conversion f32 -> bf16, bias combine
    k_cvt_bf16<<<(WG + 255) / 256, 256, 0, stream>>>(enc_Wih, encWih_b, WG);
    k_cvt_bf16<<<(WG + 255) / 256, 256, 0, stream>>>(enc_Whh, encWhh_b, WG);
    k_cvt_bf16<<<(WG + 255) / 256, 256, 0, stream>>>(dec_Wih, decWih_b, WG);
    k_cvt_bf16<<<(WG + 255) / 256, 256, 0, stream>>>(dec_Whh, decWhh_b, WG);
    k_cvt_bf16<<<(VOCAB * HID + 255) / 256, 256, 0, stream>>>(Wout, Wout_b, VOCAB * HID);
    k_bias_sum<<<(4 * HID + 255) / 256, 256, 0, stream>>>(enc_bih, enc_bhh, enc_bias, 4 * HID);
    k_bias_sum<<<(4 * HID + 255) / 256, 256, 0, stream>>>(dec_bih, dec_bhh, dec_bias, 4 * HID);

    // 2) embedding gathers (dec gather also seeds A_proj t=0 rows)
    {
        int n = S_ENC * BATCH * HID;
        k_gather<<<(n + 255) / 256, 256, 0, stream>>>(inp, enc_emb, x_enc, nullptr, S_ENC, n);
        n = T_DEC * BATCH * HID;
        k_gather<<<(n + 255) / 256, 256, 0, stream>>>(tar, dec_emb, x_dec, aproj, T_DEC, n);
    }

    // 3) zero-init h ping buffer and barrier counter (every launch -> replay-safe)
    k_zero_u16<<<(BATCH * HID + 255) / 256, 256, 0, stream>>>(hb0, BATCH * HID);
    k_zero_u32<<<1, 64, 0, stream>>>(ctr, 64);

    // 4) persistent encoder+decoder scan: one launch, 191 steps,
    //    LDS-resident weights, VGPR-resident cell state, atomic grid barrier
    k_scan<<<NBLK, 128, 0, stream>>>(x_enc, x_dec,
                                     encWih_b, encWhh_b, decWih_b, decWhh_b,
                                     enc_bias, dec_bias,
                                     hb0, hb1, aproj, ctr);

    // 5) projection + bias + relu -> f32 logits in d_out
    //    500 N-panels x 32 M-blocks = 16000 blocks of 4 waves
    k_proj<<<16000, 128, 0, stream>>>(aproj, Wout_b, bout, out);

    // 6) log-softmax over vocab (rowLSE), then over batch, fused subtraction
    k_rowlse<<<BATCH * T_DEC, 256, 0, stream>>>(out, rowlse);
    k_final<<<(T_DEC * VOCAB + 255) / 256, 256, 0, stream>>>(out, rowlse);
}
